// SelfAttention_77919296684338
// MI455X (gfx1250) — matrix operand order repaired and checked by reading, not compiled, not run on hardware
//
#include <hip/hip_runtime.h>
#include <hip/hip_bf16.h>

// ---------------------------------------------------------------------------
// Self-attention (q=k=v=X reshaped) + output projection for MI455X (gfx1250).
// All matmuls via v_wmma_f32_16x16x32_bf16 (wave32 WMMA), flash-style online
// softmax. All WMMA fragments are built from contiguous b64/b128 loads:
//   A fragment (per lane): runs [g*8, g*8+8) and [16+g*8, ...) of the row
//   B fragment (per lane): elements e <-> contraction k = g*8+e (e<8) / 16+g*8+e-8 (e>=8), contiguous
//                          when the contraction dim is the fast axis in LDS.
// A transposed copy kvT[d][kk] is staged so the P*V B-fragment is contiguous.
// ---------------------------------------------------------------------------

#define NB   2      // batch
#define LL   2048   // sequence length
#define EE   1024   // embed
#define HH   16     // heads
#define DD   64     // head dim
#define LH   72     // LDS row stride (bf16 elems), pad vs bank conflicts
#define WLH  40     // LDS row stride (bf16 elems) for W tile in projection

typedef __attribute__((ext_vector_type(16))) __bf16 v16bf;
typedef __attribute__((ext_vector_type(8)))  float  v8f;

union V16U { v16bf v; unsigned long long q[4]; };

static __device__ __forceinline__ v8f wmma_bf16(const v16bf& a, const v16bf& b, const v8f& c) {
  return __builtin_amdgcn_wmma_f32_16x16x32_bf16(
      /*neg_a=*/false, a, /*neg_b=*/false, b,
      /*c_mod=*/(short)0, c, /*reuse_a=*/false, /*reuse_b=*/false);
}

// Build a B fragment (32x16, contraction-major storage) from 8-byte chunks.
// base points at row (= fragment column n) start; element e <-> k = g*8 + e (e<8), 16 + g*8 + (e-8) (e>=8).
static __device__ __forceinline__ v16bf load_bfrag(const __bf16* rowbase, int g, int koff) {
  V16U b;
  const unsigned long long* s64 = (const unsigned long long*)(rowbase + koff);
#pragma unroll
  for (int w = 0; w < 4; ++w) b.q[w] = s64[(w >> 1) * 4 + g * 2 + (w & 1)];
  return b.v;
}

// Build an A fragment (16x32) from bf16 row-major storage: two 16B runs.
static __device__ __forceinline__ v16bf load_afrag_bf16(const __bf16* row, int g, int koff) {
  V16U a;
  const unsigned long long* s64 = (const unsigned long long*)(row + koff);
  a.q[0] = s64[g * 2];
  a.q[1] = s64[g * 2 + 1];
  a.q[2] = s64[4 + g * 2];
  a.q[3] = s64[4 + g * 2 + 1];
  return a.v;
}

// ===========================================================================
// Kernel 1: flash attention. One (n,h) per blockIdx.x, 64 Q rows per block
// (4 waves x 16 rows). Streams 64-wide K/V blocks through LDS (kv + kvT).
// Writes O (softmax(QK^T/32 + mask) V) as bf16 into workspace, layout (N,L,E).
// ===========================================================================
__global__ __launch_bounds__(128) void attn_flash_wmma(
    const float* __restrict__ X, const int* __restrict__ mask,
    __bf16* __restrict__ attn) {
  __shared__ __bf16 kv [64 * LH];       // [k_local][d]   (for Q*K^T B-frags)
  __shared__ __bf16 kvT[64 * LH];       // [d][k_local]   (for P*V  B-frags)
  __shared__ __bf16 pl[4][16 * LH];     // per-wave P tile: [q_local][k_local]

  const int nh   = blockIdx.x;          // 0..31
  const int n    = nh >> 4;
  const int h    = nh & 15;
  const int qt   = blockIdx.y;          // 0..31
  const int wave = threadIdx.x >> 5;
  const int lane = threadIdx.x & 31;
  const int g    = lane >> 4;           // lane-half
  const int ln   = lane & 15;

  const float* Xh    = X + (size_t)n * LL * EE + h * DD;   // + q*EE per row
  const int    qbase = qt * 64 + wave * 16;

  // ---- Q tile in A-matrix layout (fp32 global -> bf16), rows M = ln
  v16bf aq0, aq1;
  {
    const float* qrow = Xh + (size_t)(qbase + ln) * EE;
#pragma unroll
    for (int blk = 0; blk < 2; ++blk) {
      float4 f0 = *(const float4*)(qrow + blk * 32 + g * 8);
      float4 f1 = *(const float4*)(qrow + blk * 32 + g * 8 + 4);
      float4 f2 = *(const float4*)(qrow + blk * 32 + 16 + g * 8);
      float4 f3 = *(const float4*)(qrow + blk * 32 + 16 + g * 8 + 4);
      v16bf& a = blk ? aq1 : aq0;
      a[0]=(__bf16)f0.x; a[1]=(__bf16)f0.y; a[2]=(__bf16)f0.z; a[3]=(__bf16)f0.w;
      a[4]=(__bf16)f1.x; a[5]=(__bf16)f1.y; a[6]=(__bf16)f1.z; a[7]=(__bf16)f1.w;
      a[8]=(__bf16)f2.x; a[9]=(__bf16)f2.y; a[10]=(__bf16)f2.z; a[11]=(__bf16)f2.w;
      a[12]=(__bf16)f3.x; a[13]=(__bf16)f3.y; a[14]=(__bf16)f3.z; a[15]=(__bf16)f3.w;
    }
  }

  // ---- online softmax state (rows M = g*8 + r)
  float m[8], l[8];
  v8f   oacc[4];
#pragma unroll
  for (int r = 0; r < 8; ++r) { m[r] = -3.0e38f; l[r] = 0.0f; }
#pragma unroll
  for (int jd = 0; jd < 4; ++jd) oacc[jd] = (v8f){0,0,0,0,0,0,0,0};

  const float scale = 0.03125f;  // 1/sqrt(1024)

  for (int kb = 0; kb < LL / 64; ++kb) {
    const int kbase = kb * 64;

    __syncthreads();  // previous block fully consumed
    // ---- cooperative stage: 64 rows x 64 d, fp32 -> bf16, direct + transposed
    for (int t = threadIdx.x; t < 64 * 16; t += 128) {
      int row = t >> 4;
      int c4  = (t & 15) << 2;
      const float4 f = *(const float4*)(Xh + (size_t)(kbase + row) * EE + c4);
      __bf16 b0 = (__bf16)f.x, b1 = (__bf16)f.y, b2 = (__bf16)f.z, b3 = (__bf16)f.w;
      __bf16* dst = &kv[row * LH + c4];
      dst[0] = b0; dst[1] = b1; dst[2] = b2; dst[3] = b3;
      kvT[(c4 + 0) * LH + row] = b0;
      kvT[(c4 + 1) * LH + row] = b1;
      kvT[(c4 + 2) * LH + row] = b2;
      kvT[(c4 + 3) * LH + row] = b3;
    }
    if (kb + 1 < LL / 64) {  // emits global_prefetch_b8
      __builtin_prefetch(Xh + (size_t)(kbase + 64 + (threadIdx.x >> 1)) * EE, 0, 0);
    }
    __syncthreads();

    // ---- scores: S = Q K^T for 4 column tiles of 16 (B from kv, contiguous)
    v8f s[4];
#pragma unroll
    for (int jj = 0; jj < 4; ++jj) {
      const __bf16* rowb = &kv[(jj * 16 + ln) * LH];
      v8f c = (v8f){0,0,0,0,0,0,0,0};
      c = wmma_bf16(aq0, load_bfrag(rowb, g, 0),  c);
      c = wmma_bf16(aq1, load_bfrag(rowb, g, 32), c);
      s[jj] = c;
    }

    // ---- mask + scale (mask is per K column, shared across Q rows)
#pragma unroll
    for (int jj = 0; jj < 4; ++jj) {
      const int mv = mask[n * LL + kbase + jj * 16 + ln];
#pragma unroll
      for (int r = 0; r < 8; ++r) {
        float val = s[jj][r] * scale;
        s[jj][r] = (mv == 0) ? -1.0e20f : val;
      }
    }

    // ---- block row max (across 4 tiles, then across the 16-lane half)
    float mnew[8];
#pragma unroll
    for (int r = 0; r < 8; ++r) {
      float mx = fmaxf(fmaxf(s[0][r], s[1][r]), fmaxf(s[2][r], s[3][r]));
#pragma unroll
      for (int off = 1; off < 16; off <<= 1) mx = fmaxf(mx, __shfl_xor(mx, off, 32));
      mnew[r] = fmaxf(m[r], mx);
    }

    // ---- exponentiate, stash P (bf16) into per-wave LDS tile, row sums
    __bf16* pw = &pl[wave][0];
    float rs[8];
#pragma unroll
    for (int r = 0; r < 8; ++r) rs[r] = 0.0f;
#pragma unroll
    for (int jj = 0; jj < 4; ++jj) {
#pragma unroll
      for (int r = 0; r < 8; ++r) {
        float p = __expf(s[jj][r] - mnew[r]);
        rs[r] += p;
        pw[(g * 8 + r) * LH + jj * 16 + ln] = (__bf16)p;  // C/D layout -> [q][k]
      }
    }
#pragma unroll
    for (int r = 0; r < 8; ++r) {
      float alpha = __expf(m[r] - mnew[r]);
      float sum = rs[r];
#pragma unroll
      for (int off = 1; off < 16; off <<= 1) sum += __shfl_xor(sum, off, 32);
      l[r] = l[r] * alpha + sum;
      m[r] = mnew[r];
#pragma unroll
      for (int jd = 0; jd < 4; ++jd) oacc[jd][r] *= alpha;
    }

    // ---- P in A-matrix layout: contiguous 16B runs from per-wave LDS tile
    v16bf pa0 = load_afrag_bf16(pw + ln * LH, g, 0);
    v16bf pa1 = load_afrag_bf16(pw + ln * LH, g, 32);

    // ---- O += P V : B-frags from transposed copy kvT (contiguous)
#pragma unroll
    for (int jd = 0; jd < 4; ++jd) {
      const __bf16* rowb = &kvT[(jd * 16 + ln) * LH];
      oacc[jd] = wmma_bf16(pa0, load_bfrag(rowb, g, 0),  oacc[jd]);
      oacc[jd] = wmma_bf16(pa1, load_bfrag(rowb, g, 32), oacc[jd]);
    }
  }

  // ---- normalize and store O as bf16: attn[n, q, h*64 + d]
#pragma unroll
  for (int r = 0; r < 8; ++r) {
    float inv = 1.0f / l[r];
    int q = qbase + g * 8 + r;
    __bf16* orow = attn + ((size_t)n * LL + q) * EE + h * DD;
#pragma unroll
    for (int jd = 0; jd < 4; ++jd)
      orow[jd * 16 + ln] = (__bf16)(oacc[jd][r] * inv);
  }
}

// ===========================================================================
// Kernel 2: Y[4096,1024] = O_bf16 @ W^T + b  (Y[i,j] = sum_k O[i,k]*W[j,k])
// 64x64 output tile per workgroup (4 waves x 16 rows x 64 cols).
// ===========================================================================
__global__ __launch_bounds__(128) void proj_gemm_wmma(
    const __bf16* __restrict__ A, const float* __restrict__ W,
    const float* __restrict__ bias, float* __restrict__ Y) {
  __shared__ __bf16 wl[64 * WLH];  // [j_local 0..63][k_local 0..31]

  const int rt   = blockIdx.x;           // 64-row tile (0..63)
  const int ct   = blockIdx.y;           // 64-col tile (0..15)
  const int wave = threadIdx.x >> 5;
  const int lane = threadIdx.x & 31;
  const int g    = lane >> 4;
  const int ln   = lane & 15;
  const int rowbase = rt * 64 + wave * 16;
  const int colbase = ct * 64;

  v8f acc[4];
#pragma unroll
  for (int jj = 0; jj < 4; ++jj) acc[jj] = (v8f){0,0,0,0,0,0,0,0};

  for (int ks = 0; ks < EE; ks += 32) {
    __syncthreads();
    // stage W[colbase+j][ks..ks+31] fp32 -> bf16 LDS (j rows, k fast axis)
    for (int t = threadIdx.x; t < 64 * 8; t += 128) {
      int j  = t >> 3;
      int c4 = (t & 7) << 2;
      const float4 f = *(const float4*)(W + (size_t)(colbase + j) * EE + ks + c4);
      __bf16* dst = &wl[j * WLH + c4];
      dst[0] = (__bf16)f.x; dst[1] = (__bf16)f.y;
      dst[2] = (__bf16)f.z; dst[3] = (__bf16)f.w;
    }
    if (ks + 32 < EE) {
      __builtin_prefetch(W + (size_t)(colbase + (threadIdx.x >> 1)) * EE + ks + 32, 0, 0);
    }
    __syncthreads();

    // A fragment straight from bf16 workspace (rows M = ln), contiguous runs
    v16bf a = load_afrag_bf16(A + (size_t)(rowbase + ln) * EE + ks, g, 0);

#pragma unroll
    for (int jj = 0; jj < 4; ++jj) {
      acc[jj] = wmma_bf16(a, load_bfrag(&wl[(jj * 16 + ln) * WLH], g, 0), acc[jj]);
    }
  }

  // store: row i = rowbase + g*8 + r, col j = colbase + jj*16 + ln
#pragma unroll
  for (int r = 0; r < 8; ++r) {
    int i = rowbase + g * 8 + r;
#pragma unroll
    for (int jj = 0; jj < 4; ++jj) {
      int j = colbase + jj * 16 + ln;
      Y[(size_t)i * EE + j] = acc[jj][r] + bias[j];
    }
  }
}

// ===========================================================================
extern "C" void kernel_launch(void* const* d_in, const int* in_sizes, int n_in,
                              void* d_out, int out_size, void* d_ws, size_t ws_size,
                              hipStream_t stream) {
  const float* X    = (const float*)d_in[0];
  const int*   mask = (const int*)d_in[1];
  const float* W    = (const float*)d_in[2];
  const float* b    = (const float*)d_in[3];
  float*       Y    = (float*)d_out;
  __bf16*      attn = (__bf16*)d_ws;  // (N,L,E) bf16 = 8 MB

  dim3 g1(NB * HH, LL / 64);  // 32 x 32 workgroups
  attn_flash_wmma<<<g1, 128, 0, stream>>>(X, mask, attn);

  dim3 g2((NB * LL) / 64, EE / 64);  // 64 x 16 workgroups
  proj_gemm_wmma<<<g2, 128, 0, stream>>>(attn, W, b, Y);
}
